// CNNtoRNN_5720896438864
// MI455X (gfx1250) — matrix operand order repaired and checked by reading, not compiled, not run on hardware
//
#include <hip/hip_runtime.h>

// ---------------------------------------------------------------------------
// CDNA5 (gfx1250) CNN encoder -> attention-LSTM decoder.
//  * All matrix math on v_wmma_f32_16x16x32_f16 (f32 accum), wave32.
//  * Weights pre-swizzled once per launch into "fragment-major" f16 tiles:
//    each 32x64 K-tile is stored so one lane's WMMA B-fragment is 32
//    contiguous bytes -> B streams L2 -> VGPR with two b128 loads, no LDS.
//  * Activations are f16; GEMM A tiles are staged into LDS with
//    GLOBAL_LOAD_ASYNC_TO_LDS_B128 (ASYNCcnt) when the toolchain exposes it.
// ---------------------------------------------------------------------------

typedef __attribute__((ext_vector_type(16))) _Float16 v16h;
typedef __attribute__((ext_vector_type(8)))  float    v8f;
typedef int v4i_ __attribute__((vector_size(16)));   // builtin's b128 payload type

#define BN_EPS 1e-5f

#if defined(__AMDGCN__) && __has_builtin(__builtin_amdgcn_global_load_async_to_lds_b128) && __has_builtin(__builtin_amdgcn_s_wait_asynccnt)
#define USE_ASYNC 1
#else
#define USE_ASYNC 0
#endif

static __device__ __forceinline__ float sigmoidf_(float x) {
  return 1.0f / (1.0f + __expf(-x));
}

// 16-byte global -> LDS copy. Async (ASYNCcnt DMA) when available.
static __device__ __forceinline__ void lds_copy16(_Float16* dst, const _Float16* src) {
#if USE_ASYNC
  __builtin_amdgcn_global_load_async_to_lds_b128(
      (__attribute__((address_space(1))) v4i_*)(uintptr_t)(const void*)src,
      (__attribute__((address_space(3))) v4i_*)(uintptr_t)(void*)dst, 0, 0);
#else
  *(uint4*)dst = *(const uint4*)src;
#endif
}

static __device__ __forceinline__ void wait_async0() {
#if USE_ASYNC
  __builtin_amdgcn_s_wait_asynccnt(0);
#endif
}

// Fragment-major layout: a 32(K) x 64(N or M) tile = 4 fragments x 32 lanes x
// 16 halves; one lane's fragment data is 16 contiguous halves (32 B).
//   A (16x32, MxK): lane = (m&15) + 16*((k>>3)&1); i = (k>>4)*8 + (k&7)
//   B (32x16, KxN): lane = (n&15) + 16*((k>>3)&1); i = (k>>4)*8 + (k&7)
static __device__ __forceinline__ v16h read_frag(const _Float16* __restrict__ s, int lane) {
  union { uint4 q[2]; v16h v; } f;
  const uint4* p = (const uint4*)(s + lane * 16);
  f.q[0] = p[0];
  f.q[1] = p[1];
  return f.v;
}

// ---------------------------------------------------------------------------
// Weight pre-swizzle: src f32 [K,N] (trans=0) or [N,K] (trans=1, conv OIHW)
// -> f16 fragment-major tiles, zero-padded at K/N tails (stem K=147).
// Tile (tk,tn) lives at ((tn*KT)+tk)*2048 halves.
// ---------------------------------------------------------------------------
__global__ void k_pack_w(const float* __restrict__ src, _Float16* __restrict__ dst,
                         int K, int N, int trans) {
  const int KT = (K + 31) >> 5, NT = (N + 63) >> 6;
  const long long total = (long long)KT * NT * 1024;  // u32 pairs
  const long long idx = (long long)blockIdx.x * 256 + threadIdx.x;
  if (idx >= total) return;
  const int tile = (int)(idx >> 10);
  const int p    = (int)(idx & 1023);
  const int tk = tile % KT, tn = tile / KT;
  const int i    = (p & 7) * 2;        // even i
  const int lane = (p >> 3) & 31;
  const int fn   = p >> 8;             // fragment (16-col group)
  const int n  = tn * 64 + fn * 16 + (lane & 15);
  const int k0 = tk * 32 + (lane >> 4) * 8 + (i >> 3) * 16 + (i & 7);
  float v0 = 0.0f, v1 = 0.0f;
  if (n < N) {
    if (trans) {
      if (k0 < K)     v0 = src[(size_t)n * K + k0];
      if (k0 + 1 < K) v1 = src[(size_t)n * K + k0 + 1];
    } else {
      if (k0 < K)     v0 = src[(size_t)k0 * N + n];
      if (k0 + 1 < K) v1 = src[(size_t)(k0 + 1) * N + n];
    }
  }
  union { _Float16 h[2]; unsigned u; } pk;
  pk.h[0] = (_Float16)v0;
  pk.h[1] = (_Float16)v1;
  *(unsigned*)(dst + (size_t)tile * 2048 + fn * 512 + lane * 16 + i) = pk.u;
}

// ---------------------------------------------------------------------------
// Dense GEMM: C[M,N](ldc) = A[M,K](f16) * Bp(packed f16) [+bias][+=C][f16 out]
// flags: 1 = add bias, 2 = accumulate into f32 C, 4 = store f16
// K must be a multiple of 32 (true for all decoder GEMMs: 512/1024).
// ---------------------------------------------------------------------------
__global__ void k_gemm_wmma(const _Float16* __restrict__ A, const _Float16* __restrict__ Bp,
                            const float* __restrict__ bias, void* __restrict__ Cv,
                            int M, int N, int K, int ldc, int flags) {
  __shared__ __align__(16) _Float16 sA[2048];
  const int tid  = threadIdx.x;
  const int lane = tid & 31;
  const int wave = tid >> 5;
  const int mtile = blockIdx.x * 64;
  const int mq = (wave >> 1) * 32;
  const int nq = (wave & 1) * 32;
  const int KT = K >> 5;
  const _Float16* Btiles = Bp + (size_t)blockIdx.y * KT * 2048;

  v8f acc[2][2] = {};

  for (int tk = 0; tk < KT; ++tk) {
    const int kt = tk << 5;
    // Stage A tile (64 rows x 32 K-halves) as 256 16-byte chunks; the
    // fragment-major LDS target keeps every chunk contiguous -> async DMA.
#pragma unroll
    for (int cc = 0; cc < 2; ++cc) {
      const int c  = (tid << 1) + cc;
      const int mi = c >> 2, q = c & 3;
      const int mrow = mtile + mi;
      _Float16* dst = sA + ((mi >> 4) << 9) + (((mi & 15) + ((q & 1) << 4)) << 4) + ((q >> 1) << 3);
      if (mrow < M) {
        lds_copy16(dst, A + (size_t)mrow * K + kt + (q << 3));
      } else {
        *(uint4*)dst = make_uint4(0u, 0u, 0u, 0u);
      }
    }
    wait_async0();
    __syncthreads();

    const _Float16* bt = Btiles + (size_t)tk * 2048;
    if (tk + 1 < KT) __builtin_prefetch(bt + 2048 + lane * 64, 0, 0);

    const v16h a0 = read_frag(sA + (((mq >> 4) + 0) << 9), lane);
    const v16h a1 = read_frag(sA + (((mq >> 4) + 1) << 9), lane);
    const v16h b0 = read_frag(bt + (((nq >> 4) + 0) << 9), lane);
    const v16h b1 = read_frag(bt + (((nq >> 4) + 1) << 9), lane);
    acc[0][0] = __builtin_amdgcn_wmma_f32_16x16x32_f16(false, a0, false, b0, (short)0, acc[0][0], false, false);
    acc[0][1] = __builtin_amdgcn_wmma_f32_16x16x32_f16(false, a0, false, b1, (short)0, acc[0][1], false, false);
    acc[1][0] = __builtin_amdgcn_wmma_f32_16x16x32_f16(false, a1, false, b0, (short)0, acc[1][0], false, false);
    acc[1][1] = __builtin_amdgcn_wmma_f32_16x16x32_f16(false, a1, false, b1, (short)0, acc[1][1], false, false);
    __syncthreads();
  }

#pragma unroll
  for (int ti = 0; ti < 2; ++ti)
#pragma unroll
    for (int tj = 0; tj < 2; ++tj)
#pragma unroll
      for (int r = 0; r < 8; ++r) {
        const int row = mtile + mq + ti * 16 + (lane >> 4) * 8 + r;
        const int col = nq + tj * 16 + (lane & 15) + blockIdx.y * 64;
        if (row < M && col < N) {
          float v = acc[ti][tj][r];
          if (flags & 1) v += bias[col];
          const size_t o = (size_t)row * ldc + col;
          if (flags & 2)      { float* C = (float*)Cv; C[o] += v; }
          else if (flags & 4) { ((_Float16*)Cv)[o] = (_Float16)v; }
          else                { ((float*)Cv)[o] = v; }
        }
      }
}

// ---------------------------------------------------------------------------
// Implicit-GEMM conv, fused BN (+residual) + ReLU. A = im2col gather staged
// into fragment-major LDS (packed u32 stores); B = packed weights direct from
// global. X is float (stem) or f16 (everything else); Y/res are f16 NCHW.
// ---------------------------------------------------------------------------
template <typename TIN>
__global__ void k_conv_wmma(const TIN* __restrict__ X, const _Float16* __restrict__ Wp,
                            const float* __restrict__ gamma, const float* __restrict__ beta,
                            const float* __restrict__ mean,  const float* __restrict__ var,
                            const _Float16* __restrict__ res, _Float16* __restrict__ Y,
                            int Bn, int Cin, int H, int W,
                            int Cout, int Ho, int Wo, int stride, int pad, int kd) {
  __shared__ __align__(16) _Float16 sA[2048];
  const int tid  = threadIdx.x;
  const int lane = tid & 31;
  const int wave = tid >> 5;
  const int M = Bn * Ho * Wo;
  const int K = Cin * kd * kd;
  const int KT = (K + 31) >> 5;
  const int mtile = blockIdx.x * 64;
  const int mq = (wave >> 1) * 32;
  const int nq = (wave & 1) * 32;
  const unsigned HoWo = (unsigned)(Ho * Wo);
  const unsigned kd2  = (unsigned)(kd * kd);
  const _Float16* Btiles = Wp + (size_t)blockIdx.y * KT * 2048;

  v8f acc[2][2] = {};

  for (int tk = 0; tk < KT; ++tk) {
    const int kt = tk << 5;
    // im2col A staging: 1024 u32 pairs (two K-adjacent halves share a lane).
#pragma unroll
    for (int pp = 0; pp < 8; ++pp) {
      const int p  = tid + pp * 128;
      const int mi = p >> 4;
      const int kp = (p & 15) << 1;
      const int mrow = mtile + mi;
      float v0 = 0.0f, v1 = 0.0f;
      if (mrow < M) {
        const unsigned um = (unsigned)mrow;
        const unsigned b   = um / HoWo;
        const unsigned rem = um % HoWo;
        const int oh = (int)(rem / (unsigned)Wo);
        const int ow = (int)(rem % (unsigned)Wo);
#pragma unroll
        for (int e = 0; e < 2; ++e) {
          const int kg = kt + kp + e;
          if (kg < K) {
            const unsigned uk = (unsigned)kg;
            const unsigned ci = uk / kd2;
            const unsigned kr = uk % kd2;
            const int kh = (int)(kr / (unsigned)kd);
            const int kw = (int)(kr % (unsigned)kd);
            const int ih = oh * stride - pad + kh;
            const int iw = ow * stride - pad + kw;
            if (ih >= 0 && ih < H && iw >= 0 && iw < W) {
              const float xv = (float)X[(((size_t)b * Cin + ci) * H + ih) * W + iw];
              if (e == 0) v0 = xv; else v1 = xv;
            }
          }
        }
      }
      union { _Float16 h[2]; unsigned u; } pk;
      pk.h[0] = (_Float16)v0;
      pk.h[1] = (_Float16)v1;
      const int lane_ = (mi & 15) + (((kp >> 3) & 1) << 4);
      const int i     = ((kp >> 4) << 3) + (kp & 7);
      *(unsigned*)(sA + ((mi >> 4) << 9) + (lane_ << 4) + i) = pk.u;
    }
    __syncthreads();

    const _Float16* bt = Btiles + (size_t)tk * 2048;
    if (tk + 1 < KT) __builtin_prefetch(bt + 2048 + lane * 64, 0, 0);

    const v16h a0 = read_frag(sA + (((mq >> 4) + 0) << 9), lane);
    const v16h a1 = read_frag(sA + (((mq >> 4) + 1) << 9), lane);
    const v16h b0 = read_frag(bt + (((nq >> 4) + 0) << 9), lane);
    const v16h b1 = read_frag(bt + (((nq >> 4) + 1) << 9), lane);
    acc[0][0] = __builtin_amdgcn_wmma_f32_16x16x32_f16(false, a0, false, b0, (short)0, acc[0][0], false, false);
    acc[0][1] = __builtin_amdgcn_wmma_f32_16x16x32_f16(false, a0, false, b1, (short)0, acc[0][1], false, false);
    acc[1][0] = __builtin_amdgcn_wmma_f32_16x16x32_f16(false, a1, false, b0, (short)0, acc[1][0], false, false);
    acc[1][1] = __builtin_amdgcn_wmma_f32_16x16x32_f16(false, a1, false, b1, (short)0, acc[1][1], false, false);
    __syncthreads();
  }

#pragma unroll
  for (int ti = 0; ti < 2; ++ti)
#pragma unroll
    for (int tj = 0; tj < 2; ++tj)
#pragma unroll
      for (int r = 0; r < 8; ++r) {
        const int row = mtile + mq + ti * 16 + (lane >> 4) * 8 + r;
        const int col = blockIdx.y * 64 + nq + tj * 16 + (lane & 15);
        if (row < M && col < Cout) {
          const unsigned um = (unsigned)row;
          const unsigned b   = um / HoWo;
          const unsigned rem = um % HoWo;
          const float scale = gamma[col] * rsqrtf(var[col] + BN_EPS);
          const float shift = beta[col] - mean[col] * scale;
          float v = acc[ti][tj][r] * scale + shift;
          const size_t oidx = ((size_t)b * Cout + col) * HoWo + rem;
          if (res) v += (float)res[oidx];
          Y[oidx] = (_Float16)fmaxf(v, 0.0f);
        }
      }
}

// --------------------------- pointwise kernels -----------------------------

__global__ void k_maxpool3s2(const _Float16* __restrict__ in, _Float16* __restrict__ out,
                             int Bn, int C, int H, int W, int Ho, int Wo) {
  const int idx = blockIdx.x * blockDim.x + threadIdx.x;
  if (idx >= Bn * C * Ho * Wo) return;
  int t = idx;
  const int wo = t % Wo; t /= Wo;
  const int ho = t % Ho; t /= Ho;
  const int c  = t % C;  t /= C;
  const int b  = t;
  float m = -INFINITY;
#pragma unroll
  for (int kh = 0; kh < 3; ++kh)
#pragma unroll
    for (int kw = 0; kw < 3; ++kw) {
      const int ih = ho * 2 - 1 + kh, iw = wo * 2 - 1 + kw;
      if (ih >= 0 && ih < H && iw >= 0 && iw < W)
        m = fmaxf(m, (float)in[(((size_t)b * C + c) * H + ih) * W + iw]);
    }
  out[idx] = (_Float16)m;
}

// NCHW [B,512,7,7] f16 -> feats [B,49,512] f16
__global__ void k_nchw_to_nld(const _Float16* __restrict__ in, _Float16* __restrict__ out) {
  const int idx = blockIdx.x * blockDim.x + threadIdx.x;
  if (idx >= 32 * 512 * 49) return;
  int t = idx;
  const int p = t % 49;  t /= 49;
  const int c = t % 512; t /= 512;
  const int b = t;
  out[((size_t)b * 49 + p) * 512 + c] = in[idx];
}

__global__ void k_mean49(const _Float16* __restrict__ feats, _Float16* __restrict__ out) {
  const int idx = blockIdx.x * blockDim.x + threadIdx.x;
  if (idx >= 32 * 512) return;
  const int b = idx >> 9, d = idx & 511;
  float s = 0.0f;
  for (int n = 0; n < 49; ++n) s += (float)feats[((size_t)b * 49 + n) * 512 + d];
  out[idx] = (_Float16)(s * (1.0f / 49.0f));
}

__global__ void k_embed(const int* __restrict__ cap, const float* __restrict__ emb,
                        float* __restrict__ out) {
  const int idx = blockIdx.x * blockDim.x + threadIdx.x;
  if (idx >= 32 * 21 * 512) return;
  int t = idx;
  const int e  = t % 512; t /= 512;
  const int tt = t % 21;  t /= 21;
  const int b  = t;
  out[idx] = emb[(size_t)cap[b * 21 + tt] * 512 + e];
}

// scores[b,n] = tanh(encproj[b,n,:] + hp[b,:]) . Aw + Ab  (one wave per row)
__global__ void k_scores(const float* __restrict__ encproj, const float* __restrict__ hp,
                         const float* __restrict__ Aw, const float* __restrict__ Ab,
                         float* __restrict__ scores) {
  const int warp = threadIdx.x >> 5, lane = threadIdx.x & 31;
  const int row = blockIdx.x * 4 + warp;
  if (row >= 32 * 49) return;
  const int b = row / 49;
  float s = 0.0f;
  for (int a = lane; a < 512; a += 32)
    s += tanhf(encproj[(size_t)row * 512 + a] + hp[(size_t)b * 512 + a]) * Aw[a];
#pragma unroll
  for (int off = 16; off > 0; off >>= 1) s += __shfl_xor(s, off, 32);
  if (lane == 0) scores[row] = s + Ab[0];
}

__global__ void k_softmax49(const float* __restrict__ scores, float* __restrict__ attn) {
  __shared__ float s[64];
  const int b = blockIdx.x, t = threadIdx.x;
  const float v = (t < 49) ? scores[b * 49 + t] : -INFINITY;
  s[t] = v;
  __syncthreads();
  for (int st = 32; st > 0; st >>= 1) { if (t < st) s[t] = fmaxf(s[t], s[t + st]); __syncthreads(); }
  const float mx = s[0];
  __syncthreads();
  const float e = (t < 49) ? __expf(v - mx) : 0.0f;
  s[t] = e;
  __syncthreads();
  for (int st = 32; st > 0; st >>= 1) { if (t < st) s[t] += s[t + st]; __syncthreads(); }
  const float sum = s[0];
  if (t < 49) attn[b * 49 + t] = e / sum;
}

__global__ void k_context(const float* __restrict__ attn, const _Float16* __restrict__ feats,
                          float* __restrict__ ctx) {
  const int idx = blockIdx.x * blockDim.x + threadIdx.x;
  if (idx >= 32 * 512) return;
  const int b = idx >> 9, d = idx & 511;
  float s = 0.0f;
  for (int n = 0; n < 49; ++n)
    s += attn[b * 49 + n] * (float)feats[((size_t)b * 49 + n) * 512 + d];
  ctx[idx] = s;
}

// x = [emb_t, sigmoid(fb) * context]  (B x 1024, f16 for GEMM-A use)
__global__ void k_xcat(const float* __restrict__ emb_all, int t,
                       const float* __restrict__ fb, const float* __restrict__ ctx,
                       _Float16* __restrict__ x) {
  const int idx = blockIdx.x * blockDim.x + threadIdx.x;
  if (idx >= 32 * 1024) return;
  const int b = idx >> 10, j = idx & 1023;
  float v;
  if (j < 512) v = emb_all[((size_t)b * 21 + t) * 512 + j];
  else {
    const int jj = j - 512;
    v = sigmoidf_(fb[b * 512 + jj]) * ctx[b * 512 + jj];
  }
  x[idx] = (_Float16)v;
}

// gates [B,2048] = [i|f|g|o]; c (f32) updated in place, h stored f16
__global__ void k_lstm(const float* __restrict__ gates, _Float16* __restrict__ h,
                       float* __restrict__ c) {
  const int idx = blockIdx.x * blockDim.x + threadIdx.x;
  if (idx >= 32 * 512) return;
  const int b = idx >> 9, j = idx & 511;
  const float* g = gates + (size_t)b * 2048;
  const float gi = sigmoidf_(g[j]);
  const float gf = sigmoidf_(g[512 + j]);
  const float gg = tanhf(g[1024 + j]);
  const float go = sigmoidf_(g[1536 + j]);
  const float cn = gf * c[idx] + gi * gg;
  c[idx] = cn;
  h[idx] = (_Float16)(go * tanhf(cn));
}

// ---------------------------------------------------------------------------
extern "C" void kernel_launch(void* const* d_in, const int* in_sizes, int n_in,
                              void* d_out, int out_size, void* d_ws, size_t ws_size,
                              hipStream_t stream) {
  (void)in_sizes; (void)n_in; (void)out_size; (void)ws_size;
  auto F = [&](int i) { return (const float*)d_in[i]; };
  const int* captions = (const int*)d_in[1];

  // ---- workspace carve-out (256B-aligned slabs) ----
  float* w = (float*)d_ws;
  size_t off = 0;
  auto alloc  = [&](size_t nf) { float* p = w + off; off += (nf + 63) & ~(size_t)63; return p; };
  auto allocH = [&](size_t nh) { return (_Float16*)alloc((nh + 1) / 2); };

  _Float16* bufA  = allocH((size_t)32 * 64 * 112 * 112);
  _Float16* bufB  = allocH((size_t)32 * 64 * 56 * 56);
  _Float16* bufC  = allocH((size_t)32 * 128 * 28 * 28);
  _Float16* bufD  = allocH((size_t)32 * 128 * 28 * 28);
  _Float16* bufE  = allocH((size_t)32 * 128 * 28 * 28);
  _Float16* feats = allocH((size_t)32 * 49 * 512);
  _Float16* meanf = allocH((size_t)32 * 512);
  _Float16* h_    = allocH((size_t)32 * 512);
  _Float16* xcat  = allocH((size_t)32 * 1024);
  float* eproj = alloc((size_t)32 * 49 * 512);
  float* c_    = alloc(32 * 512);
  float* hp    = alloc(32 * 512);
  float* fb    = alloc(32 * 512);
  float* ctx   = alloc(32 * 512);
  float* gates = alloc(32 * 2048);
  float* scor  = alloc(32 * 49);
  float* attn  = alloc(32 * 49);
  float* embA  = alloc((size_t)32 * 21 * 512);

  // ---- weight pre-swizzle into fragment-major f16 tiles ----
  auto pack = [&](const float* src, int K, int N, int trans) -> const _Float16* {
    const int KT = (K + 31) >> 5, NT = (N + 63) >> 6;
    _Float16* dst = allocH((size_t)KT * NT * 2048);
    const long long pairs = (long long)KT * NT * 1024;
    k_pack_w<<<(unsigned)((pairs + 255) / 256), 256, 0, stream>>>(src, dst, K, N, trans);
    return dst;
  };
  const _Float16* pw0 = pack(F(2), 3 * 49, 64, 1);                 // stem (OIHW)
  const _Float16* pwL[3][5];
  const int bases[3] = {7, 32, 57};
  const int cis[3] = {64, 128, 256}, cos_[3] = {128, 256, 512};
  for (int l = 0; l < 3; ++l) {
    pwL[l][0] = pack(F(bases[l]),      cis[l] * 9,  cos_[l], 1);
    pwL[l][1] = pack(F(bases[l] + 5),  cos_[l] * 9, cos_[l], 1);
    pwL[l][2] = pack(F(bases[l] + 10), cos_[l] * 9, cos_[l], 1);
    pwL[l][3] = pack(F(bases[l] + 15), cos_[l] * 9, cos_[l], 1);
    pwL[l][4] = pack(F(bases[l] + 20), cos_[l] * 9, cos_[l], 1);
  }
  const _Float16* pU   = pack(F(82), 512,  512,   0);
  const _Float16* pW   = pack(F(84), 512,  512,   0);
  const _Float16* pIH  = pack(F(89), 512,  512,   0);
  const _Float16* pIC  = pack(F(91), 512,  512,   0);
  const _Float16* pWih = pack(F(93), 1024, 2048,  0);
  const _Float16* pWhh = pack(F(95), 512,  2048,  0);
  const _Float16* pFb  = pack(F(97), 512,  512,   0);
  const _Float16* pFc  = pack(F(99), 512,  10000, 0);

  auto conv = [&](auto X, const _Float16* pw, int bni, const _Float16* res, _Float16* y,
                  int Ci, int Hi, int Co, int Ho, int s, int p, int kd) {
    const int M = 32 * Ho * Ho;
    dim3 g((M + 63) / 64, (Co + 63) / 64), blk(128);
    k_conv_wmma<<<g, blk, 0, stream>>>(X, pw, F(bni), F(bni + 1), F(bni + 2), F(bni + 3),
                                       res, y, 32, Ci, Hi, Hi, Co, Ho, Ho, s, p, kd);
  };
  auto gemm = [&](const _Float16* A, const _Float16* Bp, const float* bias, void* C,
                  int M, int N, int K, int ldc, int flags) {
    dim3 g((M + 63) / 64, (N + 63) / 64), blk(128);
    k_gemm_wmma<<<g, blk, 0, stream>>>(A, Bp, bias, C, M, N, K, ldc, flags);
  };
  auto layer = [&](const _Float16* x, int l, int Ci, int Hi, int Co,
                   _Float16* t0, _Float16* t1, _Float16* t2) {
    const int base = bases[l];
    const int Ho = Hi / 2;
    conv(x,  pwL[l][0], base + 1,  (const _Float16*)nullptr, t0, Ci, Hi, Co, Ho, 2, 1, 3);
    conv(t0, pwL[l][1], base + 6,  (const _Float16*)nullptr, t1, Co, Ho, Co, Ho, 1, 1, 3);
    conv(t1, pwL[l][2], base + 11, t0,                        t2, Co, Ho, Co, Ho, 1, 1, 3);
    conv(t2, pwL[l][3], base + 16, (const _Float16*)nullptr,  t1, Co, Ho, Co, Ho, 1, 1, 3);
    conv(t1, pwL[l][4], base + 21, t2,                        t0, Co, Ho, Co, Ho, 1, 1, 3);
  };

  // ---------------- encoder ----------------
  conv(F(0), pw0, 3, (const _Float16*)nullptr, bufA, 3, 224, 64, 112, 2, 3, 7);
  {
    const int total = 32 * 64 * 56 * 56;
    k_maxpool3s2<<<(total + 255) / 256, 256, 0, stream>>>(bufA, bufB, 32, 64, 112, 112, 56, 56);
  }
  layer(bufB, 0, 64,  56, 128, bufC, bufD, bufE);   // -> bufC
  layer(bufC, 1, 128, 28, 256, bufD, bufE, bufB);   // -> bufD
  layer(bufD, 2, 256, 14, 512, bufC, bufE, bufB);   // -> bufC [32,512,7,7]

  // ---------------- decoder ----------------
  k_nchw_to_nld<<<(32 * 512 * 49 + 255) / 256, 256, 0, stream>>>(bufC, feats);
  k_mean49<<<(32 * 512 + 255) / 256, 256, 0, stream>>>(feats, meanf);
  k_embed<<<(32 * 21 * 512 + 255) / 256, 256, 0, stream>>>(captions, F(88), embA);

  gemm(meanf, pIH, F(90), h_,    32,   512, 512, 512, 1 | 4);   // init_h (f16)
  gemm(meanf, pIC, F(92), c_,    32,   512, 512, 512, 1);       // init_c (f32)
  gemm(feats, pU,  F(83), eproj, 1568, 512, 512, 512, 1);       // U projection

  float* out = (float*)d_out;                                   // [32, 20, 10000]
  for (int t = 0; t < 20; ++t) {
    gemm(h_, pW, F(85), hp, 32, 512, 512, 512, 1);              // W h
    k_scores<<<(32 * 49 + 3) / 4, 128, 0, stream>>>(eproj, hp, F(86), F(87), scor);
    k_softmax49<<<32, 64, 0, stream>>>(scor, attn);
    k_context<<<(32 * 512 + 255) / 256, 256, 0, stream>>>(attn, feats, ctx);
    gemm(h_, pFb, F(98), fb, 32, 512, 512, 512, 1);             // f_beta
    k_xcat<<<(32 * 1024 + 255) / 256, 256, 0, stream>>>(embA, t, fb, ctx, xcat);
    gemm(xcat, pWih, F(94), gates, 32, 2048, 1024, 2048, 1);    // x @ W_ih + b_ih
    gemm(h_,   pWhh, F(96), gates, 32, 2048, 512,  2048, 3);    // += h @ W_hh + b_hh
    k_lstm<<<(32 * 512 + 255) / 256, 256, 0, stream>>>(gates, h_, c_);
    gemm(h_, pFc, F(100), out + (size_t)t * 10000, 32, 10000, 512, 20 * 10000, 1);
  }
}
